// P2BVoteNetRPN_50199577756255
// MI455X (gfx1250) — matrix operand order repaired and checked by reading, not compiled, not run on hardware
//
#include <hip/hip_runtime.h>
#include <hip/hip_bf16.h>
#include <math.h>

// ---------------------------------------------------------------------------
// P2B VoteNet RPN for MI455X (gfx1250, wave32, WMMA).
// Heavy GEMMs: wave-tiled v_wmma_f32_16x16x32_bf16, 16x32 output per wave
// (2 accumulators share one A fragment). Strides that are 512 everywhere are
// compile-time so B-fragment loads are immediate-offset clauses; row-major A
// loads are float4 (global_load_b128).
// ---------------------------------------------------------------------------

#define Fdim 256
#define Bdim 8
#define Ndim 512
#define Sdim 64
#define EPS_BN 1e-5f
#define LDG 512                       // fixed row stride for X / Y / gram-A

typedef __attribute__((ext_vector_type(16))) __bf16 v16bf;
typedef __attribute__((ext_vector_type(8)))  float  v8f;

static __device__ __forceinline__ __bf16 f2bf(float f) { return (__bf16)f; }

// ---------------------------------------------------------------------------
// Y[b, m, n] = epilogue( sum_k A[b?, m, k] * X[b, k, n] )
//   ACM=0: A row-major [M,K], row stride ldw (AVEC=1 -> ldw%4==0, float4 loads)
//   ACM=1: A column-major with stride LDG (Gram: A[m,k] at k*LDG+m)
//   X: [B, K, N] fp32, row stride LDG.  Y: [B, M, N], row stride LDG.
//   Epilogue: +bias[m]; BN(m); *1/(1e-9+colsum[b,n]); relu; +addsrc[b,m,n].
// One wave -> 16x32 tile. M%16==0, N%32==0, K%32==0 at all call sites, so
// EXEC is all-ones at every WMMA. Fragment layouts per CDNA5 ISA 7.12.2.
// ---------------------------------------------------------------------------
template<bool ACM, bool AVEC>
__global__ void __launch_bounds__(256)
wmma_gemm(const float* __restrict__ W, int ldw, long wbs,
          const float* __restrict__ X, long xbs,
          float* __restrict__ Y, long ybs,
          int M, int N, int K,
          const float* __restrict__ bias,
          const float* __restrict__ g,  const float* __restrict__ be,
          const float* __restrict__ mu, const float* __restrict__ var,
          int relu,
          const float* __restrict__ addsrc, long addbs,
          const float* __restrict__ colsum)
{
    const int b       = blockIdx.y;
    const int tiles_n = N >> 5;                 // 32-wide column tiles
    const int wave    = threadIdx.x >> 5;
    const int tile    = blockIdx.x * 8 + wave;
    if (tile >= (M >> 4) * tiles_n) return;     // whole-wave exit only

    const int tm   = tile / tiles_n;
    const int tn   = tile - tm * tiles_n;
    const int lane = threadIdx.x & 31;
    const int half = lane >> 4;                 // 0 or 1
    const int l15  = lane & 15;
    const int arow = tm * 16 + l15;

    const float* Wb = W + (long)b * wbs;
    // A base pointer: advances by ASTEP floats per 32-deep K step.
    const float* ap = ACM ? (Wb + arow) : (Wb + (long)arow * ldw);
    const long   astep = ACM ? (long)32 * LDG : 32;
    // B base: column group start; immediate offsets cover the 32x32 K-window.
    const float* xp = X + (long)b * xbs + tn * 32 + l15;

    v8f acc0 = {}, acc1 = {};
    for (int k0 = 0; k0 < K; k0 += 32, ap += astep, xp += (long)32 * LDG) {
        v16bf afr, bf0, bf1;
        // A fragment: lane half 0 -> K {0..7, 16..23}; half 1 -> {8..15, 24..31}
        if (ACM) {
#pragma unroll
            for (int j = 0; j < 8; ++j) {
                afr[j]     = f2bf(ap[(half * 8 + j) * LDG]);
                afr[8 + j] = f2bf(ap[(16 + half * 8 + j) * LDG]);
            }
        } else if (AVEC) {
            const float4 a0 = *(const float4*)(ap + half * 8);
            const float4 a1 = *(const float4*)(ap + half * 8 + 4);
            const float4 a2 = *(const float4*)(ap + half * 8 + 16);
            const float4 a3 = *(const float4*)(ap + half * 8 + 20);
            afr[0] = f2bf(a0.x); afr[1] = f2bf(a0.y); afr[2]  = f2bf(a0.z); afr[3]  = f2bf(a0.w);
            afr[4] = f2bf(a1.x); afr[5] = f2bf(a1.y); afr[6]  = f2bf(a1.z); afr[7]  = f2bf(a1.w);
            afr[8] = f2bf(a2.x); afr[9] = f2bf(a2.y); afr[10] = f2bf(a2.z); afr[11] = f2bf(a2.w);
            afr[12] = f2bf(a3.x); afr[13] = f2bf(a3.y); afr[14] = f2bf(a3.z); afr[15] = f2bf(a3.w);
        } else {
#pragma unroll
            for (int j = 0; j < 8; ++j) {
                afr[j]     = f2bf(ap[half * 8 + j]);
                afr[8 + j] = f2bf(ap[16 + half * 8 + j]);
            }
        }
        // B fragments: lane half 0 -> K {0..7, 16..23}, half 1 -> K {8..15, 24..31};
        // two column groups 16 apart share the A fragment.
#pragma unroll
        for (int j = 0; j < 16; ++j) {
            const int off = ((j >> 3) * 16 + half * 8 + (j & 7)) * LDG;
            bf0[j] = f2bf(xp[off]);
            bf1[j] = f2bf(xp[off + 16]);
        }
        acc0 = __builtin_amdgcn_wmma_f32_16x16x32_bf16(
                   false, afr, false, bf0, (short)0, acc0, false, false);
        acc1 = __builtin_amdgcn_wmma_f32_16x16x32_bf16(
                   false, afr, false, bf1, (short)0, acc1, false, false);
    }

    // Epilogue. C/D layout: col n = l15 (+16); rows m = tm*16 + half*8 + r.
    const int nc0 = tn * 32 + l15;
    float cs0 = 1.0f, cs1 = 1.0f;
    if (colsum) {
        cs0 = 1.0f / (1e-9f + colsum[(long)b * Ndim + nc0]);
        cs1 = 1.0f / (1e-9f + colsum[(long)b * Ndim + nc0 + 16]);
    }
#pragma unroll
    for (int r = 0; r < 8; ++r) {
        const int m = tm * 16 + half * 8 + r;
        float v0 = acc0[r], v1 = acc1[r];
        if (bias) { const float t = bias[m]; v0 += t; v1 += t; }
        if (g) {
            const float s = g[m] * rsqrtf(var[m] + EPS_BN);
            const float t = be[m] - mu[m] * s;
            v0 = v0 * s + t; v1 = v1 * s + t;
        }
        v0 *= cs0; v1 *= cs1;
        if (relu) { v0 = fmaxf(v0, 0.0f); v1 = fmaxf(v1, 0.0f); }
        if (addsrc) {
            const float* a = addsrc + (long)b * addbs + (long)m * LDG + nc0;
            v0 += a[0]; v1 += a[16];
        }
        float* y = Y + (long)b * ybs + (long)m * LDG + nc0;
        y[0] = v0; y[16] = v1;
    }
}

// ------------------------- small helper kernels ----------------------------

// vote conv0 finish: y = relu(bn0(feat_part + W0[:, :3] @ (xyz - center)))
__global__ void vote_finish(const float* __restrict__ fp,
                            const float* __restrict__ w0,      // [256,259]
                            const float* __restrict__ xyz,     // [B,N,3]
                            const float* __restrict__ centers, int cstride,
                            const float* __restrict__ g,  const float* __restrict__ be,
                            const float* __restrict__ mu, const float* __restrict__ var,
                            float* __restrict__ y, long tot)
{
    long i = (long)blockIdx.x * blockDim.x + threadIdx.x;
    if (i >= tot) return;
    const int n = (int)(i % Ndim);
    const long t = i / Ndim;
    const int m = (int)(t % Fdim);
    const int b = (int)(t / Fdim);
    float acc = fp[i];
#pragma unroll
    for (int c = 0; c < 3; ++c)
        acc += w0[(long)m * (Fdim + 3) + c] *
               (xyz[((long)b * Ndim + n) * 3 + c] - centers[b * cstride + c]);
    const float s = g[m] * rsqrtf(var[m] + EPS_BN);
    acc = (acc - mu[m]) * s + be[m];
    y[i] = fmaxf(acc, 0.0f);
}

__global__ void eca_mean(const float* __restrict__ x, float* __restrict__ mean)
{
    int i = blockIdx.x * blockDim.x + threadIdx.x;
    if (i >= Bdim * Ndim) return;
    const int b = i / Ndim, n = i % Ndim;
    float s = 0.0f;
    for (int c = 0; c < Fdim; ++c)
        s += x[((long)b * Fdim + c) * Ndim + n];
    mean[i] = s * (1.0f / Fdim);
}

__global__ void eca_apply(float* __restrict__ x, const float* __restrict__ mean,
                          const float* __restrict__ w, long tot)
{
    long i = (long)blockIdx.x * blockDim.x + threadIdx.x;
    if (i >= tot) return;
    const int n = (int)(i % Ndim);
    const int b = (int)(i / ((long)Fdim * Ndim));
    const float* mb = mean + (long)b * Ndim;
    float y = w[1] * mb[n];
    if (n > 0)        y += w[0] * mb[n - 1];
    if (n < Ndim - 1) y += w[2] * mb[n + 1];
    x[i] *= 1.0f / (1.0f + __expf(-y));
}

// wave32-per-row softmax over last axis of att [B,N,N]
__global__ void softmax_rows(float* __restrict__ att, int rows)
{
    const int row  = blockIdx.x * (blockDim.x >> 5) + (threadIdx.x >> 5);
    const int lane = threadIdx.x & 31;
    if (row >= rows) return;
    float* p = att + (long)row * Ndim;
    float mx = -1e30f;
    for (int j = lane; j < Ndim; j += 32) mx = fmaxf(mx, p[j]);
#pragma unroll
    for (int o = 16; o >= 1; o >>= 1) mx = fmaxf(mx, __shfl_xor(mx, o, 32));
    float sum = 0.0f;
    for (int j = lane; j < Ndim; j += 32) { float e = __expf(p[j] - mx); p[j] = e; sum += e; }
#pragma unroll
    for (int o = 16; o >= 1; o >>= 1) sum += __shfl_xor(sum, o, 32);
    const float inv = 1.0f / sum;
    for (int j = lane; j < Ndim; j += 32) p[j] *= inv;
}

// colsum[b,j] = sum_i att[b,i,j]  (renorm denominator; folded into xr GEMM)
__global__ void col_sum(const float* __restrict__ att, float* __restrict__ cs)
{
    int i = blockIdx.x * blockDim.x + threadIdx.x;
    if (i >= Bdim * Ndim) return;
    const int b = i / Ndim, j = i % Ndim;
    const float* p = att + (long)b * Ndim * Ndim + j;
    float s = 0.0f;
    for (int r = 0; r < Ndim; ++r) s += p[(long)r * Ndim];
    cs[i] = s;
}

__global__ void elem_sub(const float* __restrict__ a, const float* __restrict__ c,
                         float* __restrict__ y, long tot)
{
    long i = (long)blockIdx.x * blockDim.x + threadIdx.x;
    if (i < tot) y[i] = a[i] - c[i];
}

__global__ void maxpool_n(const float* __restrict__ x, float* __restrict__ y, long tot)
{
    long i = (long)blockIdx.x * blockDim.x + threadIdx.x;
    if (i >= tot) return;
    const float* p = x + i * (long)Ndim;
    float m = -1e30f;
    for (int n = 0; n < Ndim; ++n) m = fmaxf(m, p[n]);
    y[i] = m;
}

// small GEMV with optional bias / BN / relu:  y[b,m] = op(W[m,:] . x[b,:])
__global__ void gemv_small(const float* __restrict__ W, int ldw,
                           const float* __restrict__ x, float* __restrict__ y,
                           int M, int K,
                           const float* __restrict__ bias,
                           const float* __restrict__ g,  const float* __restrict__ be,
                           const float* __restrict__ mu, const float* __restrict__ var,
                           int relu)
{
    int i = blockIdx.x * blockDim.x + threadIdx.x;
    if (i >= Bdim * M) return;
    const int b = i / M, m = i % M;
    const float* xb = x + (long)b * K;
    float acc = 0.0f;
    for (int k = 0; k < K; ++k) acc += W[(long)m * ldw + k] * xb[k];
    if (bias) acc += bias[m];
    if (g) { const float s = g[m] * rsqrtf(var[m] + EPS_BN); acc = (acc - mu[m]) * s + be[m]; }
    if (relu) acc = fmaxf(acc, 0.0f);
    y[i] = acc;
}

// cla head: out[b,n] = b2 + sum_k W2[k] * X[b,k,n]
__global__ void cla_last(const float* __restrict__ W2, const float* __restrict__ b2,
                         const float* __restrict__ X, float* __restrict__ out)
{
    int i = blockIdx.x * blockDim.x + threadIdx.x;
    if (i >= Bdim * Ndim) return;
    const int b = i / Ndim, n = i % Ndim;
    float acc = b2[0];
    for (int k = 0; k < Fdim; ++k)
        acc += W2[k] * X[((long)b * Fdim + k) * Ndim + n];
    out[i] = acc;
}

__global__ void box_write(const float* __restrict__ prop,
                          const float* __restrict__ centers, int cstride,
                          float* __restrict__ out, int row0)
{
    int b = threadIdx.x;
    if (b >= Bdim) return;
    float* o = out + (long)(row0 + b) * 4;
    o[0] = prop[b * 4 + 0] + centers[b * cstride + 0];
    o[1] = prop[b * 4 + 1] + centers[b * cstride + 1];
    o[2] = prop[b * 4 + 2] + centers[b * cstride + 2];
    o[3] = prop[b * 4 + 3];
}

// ------------------------------ host side ----------------------------------

struct BN4 { const float *g, *b, *m, *v; };

static inline void launch_gemm(hipStream_t st, int acm, int avec,
    const float* W, int ldw, long wbs,
    const float* X, long xbs,
    float* Y, long ybs,
    int M, int N, int K,
    const float* bias, BN4 bn, int relu,
    const float* add, long addbs,
    const float* colsum)
{
    const int tiles = (M / 16) * (N / 32);
    dim3 grid((tiles + 7) / 8, Bdim);
    if (acm)
        wmma_gemm<true, false><<<grid, 256, 0, st>>>(W, ldw, wbs, X, xbs, Y, ybs,
            M, N, K, bias, bn.g, bn.b, bn.m, bn.v, relu, add, addbs, colsum);
    else if (avec)
        wmma_gemm<false, true><<<grid, 256, 0, st>>>(W, ldw, wbs, X, xbs, Y, ybs,
            M, N, K, bias, bn.g, bn.b, bn.m, bn.v, relu, add, addbs, colsum);
    else
        wmma_gemm<false, false><<<grid, 256, 0, st>>>(W, ldw, wbs, X, xbs, Y, ybs,
            M, N, K, bias, bn.g, bn.b, bn.m, bn.v, relu, add, addbs, colsum);
}

static inline long cdiv(long a, long b) { return (a + b - 1) / b; }

extern "C" void kernel_launch(void* const* d_in, const int* in_sizes, int n_in,
                              void* d_out, int out_size, void* d_ws, size_t ws_size,
                              hipStream_t stream)
{
    // ---- inputs (top-level dict insertion order; params flattened as JAX
    // pytree leaves: nested dict keys sorted, tuples in order) ----
    const float* search_xyz   = (const float*)d_in[0];   // [B,N,3]
    const float* search_feat  = (const float*)d_in[1];   // [B,F,N]
    const float* previous_xyz = (const float*)d_in[2];   // [B,1,3]
    const float* samples      = (const float*)d_in[5];   // [B,S,3]
    int p = 6;
    const float* cla_b2 = (const float*)d_in[p++];
    BN4 cla_bn0 = {(const float*)d_in[p], (const float*)d_in[p+1], (const float*)d_in[p+2], (const float*)d_in[p+3]}; p += 4;
    BN4 cla_bn1 = {(const float*)d_in[p], (const float*)d_in[p+1], (const float*)d_in[p+2], (const float*)d_in[p+3]}; p += 4;
    const float* cla_w0 = (const float*)d_in[p++];
    const float* cla_w1 = (const float*)d_in[p++];
    const float* cla_w2 = (const float*)d_in[p++];
    const float* eca_w  = (const float*)d_in[p++];
    const float* final_b = (const float*)d_in[p++];
    const float* final_w = (const float*)d_in[p++];
    const float* prop_b2 = (const float*)d_in[p++];
    BN4 prop_bn0 = {(const float*)d_in[p], (const float*)d_in[p+1], (const float*)d_in[p+2], (const float*)d_in[p+3]}; p += 4;
    BN4 prop_bn1 = {(const float*)d_in[p], (const float*)d_in[p+1], (const float*)d_in[p+2], (const float*)d_in[p+3]}; p += 4;
    const float* prop_w0 = (const float*)d_in[p++];
    const float* prop_w1 = (const float*)d_in[p++];
    const float* prop_w2 = (const float*)d_in[p++];
    BN4 sa_bn = {(const float*)d_in[p], (const float*)d_in[p+1], (const float*)d_in[p+2], (const float*)d_in[p+3]}; p += 4;
    const float* sa_qk_w    = (const float*)d_in[p++];   // [64,256]
    const float* sa_trans_b = (const float*)d_in[p++];
    const float* sa_trans_w = (const float*)d_in[p++];
    const float* sa_v_b     = (const float*)d_in[p++];
    const float* sa_v_w     = (const float*)d_in[p++];
    const float* vote_b2 = (const float*)d_in[p++];
    BN4 vote_bn0 = {(const float*)d_in[p], (const float*)d_in[p+1], (const float*)d_in[p+2], (const float*)d_in[p+3]}; p += 4;
    BN4 vote_bn1 = {(const float*)d_in[p], (const float*)d_in[p+1], (const float*)d_in[p+2], (const float*)d_in[p+3]}; p += 4;
    const float* vote_w0 = (const float*)d_in[p++];      // [256,259]
    const float* vote_w1 = (const float*)d_in[p++];
    const float* vote_w2 = (const float*)d_in[p++];

    const BN4 no_bn = {nullptr, nullptr, nullptr, nullptr};
    float* out = (float*)d_out;                 // boxes[2080] then cla[4096]
    float* cla_out = out + (Sdim + 1) * Bdim * 4;

    // ---- workspace carve (floats) ----
    const long FN = (long)Fdim * Ndim;          // 131072
    float* ws = (float*)d_ws;
    float* featpart = ws;                       long o = Bdim * FN;
    float* x0   = ws + o;  o += Bdim * FN;
    float* x1   = ws + o;  o += Bdim * FN;
    float* xr   = ws + o;  o += Bdim * FN;
    float* kf   = ws + o;  o += (long)Bdim * 64 * Ndim;
    float* att  = ws + o;  o += (long)Bdim * Ndim * Ndim;
    float* xv   = ws + o;  o += Bdim * FN;
    float* meanb = ws + o; o += Bdim * Ndim;
    float* colsb = ws + o; o += Bdim * Ndim;
    float* pooled = ws + o; o += Bdim * Fdim;
    float* vec1 = ws + o;  o += Bdim * Fdim;
    float* vec2 = ws + o;  o += Bdim * Fdim;
    float* propb = ws + o; o += Bdim * 4;
    (void)ws_size; (void)in_sizes; (void)n_in; (void)out_size;

    const long totBFN = Bdim * FN;
    const int TB = 256;

    // =================== cla head (once) ===================
    launch_gemm(stream, 0, 1, cla_w0, Fdim, 0, search_feat, FN, x0, FN,
                Fdim, Ndim, Fdim, nullptr, cla_bn0, 1, nullptr, 0, nullptr);
    launch_gemm(stream, 0, 1, cla_w1, Fdim, 0, x0, FN, x1, FN,
                Fdim, Ndim, Fdim, nullptr, cla_bn1, 1, nullptr, 0, nullptr);
    cla_last<<<cdiv(Bdim * Ndim, TB), TB, 0, stream>>>(cla_w2, cla_b2, x1, cla_out);

    // ====== center-independent part of vote conv0 (once) ======
    // featpart = vote_w0[:, 3:] @ search_feature  (ldw=259 -> scalar A loads)
    launch_gemm(stream, 0, 0, vote_w0 + 3, Fdim + 3, 0, search_feat, FN, featpart, FN,
                Fdim, Ndim, Fdim, nullptr, no_bn, 0, nullptr, 0, nullptr);

    // =================== per-center pipeline ===================
    for (int s = 0; s <= Sdim; ++s) {
        const float* centers;
        int cstride, row0;
        if (s < Sdim) { centers = samples + s * 3; cstride = Sdim * 3; row0 = s * Bdim; }
        else          { centers = previous_xyz;    cstride = 3;        row0 = Sdim * Bdim; }

        // vote conv0 (offset contribution + bn0 + relu) -> x0
        vote_finish<<<cdiv(totBFN, TB), TB, 0, stream>>>(
            featpart, vote_w0, search_xyz, centers, cstride,
            vote_bn0.g, vote_bn0.b, vote_bn0.m, vote_bn0.v, x0, totBFN);

        // vote conv1 -> x1 ; vote conv2 (+bias) -> x0
        launch_gemm(stream, 0, 1, vote_w1, Fdim, 0, x0, FN, x1, FN,
                    Fdim, Ndim, Fdim, nullptr, vote_bn1, 1, nullptr, 0, nullptr);
        launch_gemm(stream, 0, 1, vote_w2, Fdim, 0, x1, FN, x0, FN,
                    Fdim, Ndim, Fdim, vote_b2, no_bn, 0, nullptr, 0, nullptr);

        // ECA gate (in-place on x0)
        eca_mean<<<cdiv(Bdim * Ndim, TB), TB, 0, stream>>>(x0, meanb);
        eca_apply<<<cdiv(totBFN, TB), TB, 0, stream>>>(x0, meanb, eca_w, totBFN);

        // self-attention
        // kf = qk_w @ x0                [B,64,N]
        launch_gemm(stream, 0, 1, sa_qk_w, Fdim, 0, x0, FN, kf, (long)64 * Ndim,
                    64, Ndim, Fdim, nullptr, no_bn, 0, nullptr, 0, nullptr);
        // att = kf^T @ kf  (Gram, A column-major with stride LDG)   K=64
        launch_gemm(stream, 1, 0, kf, Ndim, (long)64 * Ndim, kf, (long)64 * Ndim,
                    att, (long)Ndim * Ndim, Ndim, Ndim, 64,
                    nullptr, no_bn, 0, nullptr, 0, nullptr);
        // softmax over last axis, then column-sum for post-softmax renorm
        softmax_rows<<<cdiv((long)Bdim * Ndim * 32, TB), TB, 0, stream>>>(att, Bdim * Ndim);
        col_sum<<<cdiv(Bdim * Ndim, TB), TB, 0, stream>>>(att, colsb);
        // xv = v_w @ x0 + v_b
        launch_gemm(stream, 0, 1, sa_v_w, Fdim, 0, x0, FN, xv, FN,
                    Fdim, Ndim, Fdim, sa_v_b, no_bn, 0, nullptr, 0, nullptr);
        // xr = xv @ att, columns scaled by 1/(1e-9+colsum)   (K=512, A=xv ld 512)
        launch_gemm(stream, 0, 1, xv, Ndim, FN, att, (long)Ndim * Ndim, xr, FN,
                    Fdim, Ndim, Ndim, nullptr, no_bn, 0, nullptr, 0, colsb);
        // x1 = x0 - xr ; xr = x0 + relu(bn(trans_w @ x1 + trans_b))
        elem_sub<<<cdiv(totBFN, TB), TB, 0, stream>>>(x0, xr, x1, totBFN);
        launch_gemm(stream, 0, 1, sa_trans_w, Fdim, 0, x1, FN, xr, FN,
                    Fdim, Ndim, Fdim, sa_trans_b, sa_bn, 1, x0, FN, nullptr);

        // max-pool over N, final conv, prop head, box
        maxpool_n<<<cdiv((long)Bdim * Fdim, TB), TB, 0, stream>>>(xr, pooled, (long)Bdim * Fdim);
        gemv_small<<<cdiv(Bdim * Fdim, TB), TB, 0, stream>>>(
            final_w, Fdim, pooled, vec1, Fdim, Fdim, final_b,
            nullptr, nullptr, nullptr, nullptr, 0);
        gemv_small<<<cdiv(Bdim * Fdim, TB), TB, 0, stream>>>(
            prop_w0, Fdim, vec1, vec2, Fdim, Fdim, nullptr,
            prop_bn0.g, prop_bn0.b, prop_bn0.m, prop_bn0.v, 1);
        gemv_small<<<cdiv(Bdim * Fdim, TB), TB, 0, stream>>>(
            prop_w1, Fdim, vec2, vec1, Fdim, Fdim, nullptr,
            prop_bn1.g, prop_bn1.b, prop_bn1.m, prop_bn1.v, 1);
        gemv_small<<<1, 64, 0, stream>>>(
            prop_w2, Fdim, vec1, propb, 4, Fdim, prop_b2,
            nullptr, nullptr, nullptr, nullptr, 0);
        box_write<<<1, Bdim, 0, stream>>>(propb, centers, cstride, out, row0);
    }
}
